// SignLanguageDecoder_62156766708226
// MI455X (gfx1250) — compile-verified, matrix operand order checked by reading, not run on hardware
//
#include <hip/hip_runtime.h>
#include <stdint.h>
#include <math.h>

typedef __bf16 v16bf __attribute__((ext_vector_type(16)));
typedef float  v8f   __attribute__((ext_vector_type(8)));
typedef unsigned short u16;
typedef unsigned int   u32;
typedef unsigned long long u64;
typedef long long ll;

static __device__ __forceinline__ u16 f2bf(float f) {
  u32 u = __float_as_uint(f);
  u += 0x7FFFu + ((u >> 16) & 1u);      // round-to-nearest-even
  return (u16)(u >> 16);
}
static __device__ __forceinline__ u32 pack2(float a, float b) {
  return (u32)f2bf(a) | ((u32)f2bf(b) << 16);
}
static __device__ __forceinline__ float gelu_tanh(float v) {
  return 0.5f * v * (1.f + tanhf(0.7978845608028654f * (v + 0.044715f * v * v * v)));
}

union FragU { uint4 q[2]; u32 u[8]; v16bf v; };

static __device__ __forceinline__ v8f wmma_bf16(v16bf a, v16bf b, v8f c) {
  return __builtin_amdgcn_wmma_f32_16x16x32_bf16(false, a, false, b, (short)0, c,
                                                 false, false);
}

// CDNA5 async copy global -> LDS, 16 bytes per lane (ASYNCcnt-tracked)
static __device__ __forceinline__ void async_cp16(u32 ldsOff, const u16* g) {
  asm volatile("global_load_async_to_lds_b128 %0, %1, off"
               :: "v"(ldsOff), "v"((u64)(uintptr_t)g) : "memory");
}
static __device__ __forceinline__ void wait_async0() {
  asm volatile("s_wait_asynccnt 0" ::: "memory");
}

#define BM 128
#define BN 128
#define BK 32
#define LROW 40                       // u16 per LDS row = 80 B (16B-aligned stride)

// C[M,N] = epilogue(scale * A @ B^T_layout), bf16 operands.
// A: (M x Kd) row-major, lda.   B: "BT layout" (Ncols x Kd) row-major, ldb.
// Outputs: f32 C (opt residual-add) and/or bf16 Cb (normal [m][n] or transposed [n][m]).
// z -> (bb=z/zdiv, hh=z%zdiv); per-operand offsets bb*s?b + hh*s?h (elements).
__global__ __launch_bounds__(256) void gemm_wmma_kernel(
    const u16* __restrict__ Ag, const u16* __restrict__ Bg,
    const float* __restrict__ bias,
    float* __restrict__ Cg, u16* __restrict__ Cbg,
    int Ncols, int Kd, int lda, int ldb, int ldc, int ldcb,
    float scale, int doGelu, int doResid, int outF32, int outBF, int storeT,
    int zdiv,
    ll sAb, ll sAh, ll sBb, ll sBh, ll sCb, ll sCh, ll sCBb, ll sCBh)
{
  __shared__ __align__(16) u16 As[2][BM][LROW];
  __shared__ __align__(16) u16 Bs[2][BN][LROW];
  constexpr u32 kStageBytes = BM * LROW * 2;   // 10240 B per stage buffer
  constexpr u32 kRowBytes = LROW * 2;          // 80 B

  const int z = blockIdx.z;
  const int bb = z / zdiv, hh = z - bb * zdiv;
  const u16* A = Ag + bb * sAb + hh * sAh;
  const u16* B = Bg + bb * sBb + hh * sBh;

  const int m0 = blockIdx.y * BM;
  const int n0 = blockIdx.x * BN;
  const int tid = threadIdx.x;
  const int wave = tid >> 5, lane = tid & 31;
  const int wm = wave >> 2, wn = wave & 3;     // 2x4 wave grid, wave tile 64x32
  const int lm = lane & 15, lh = lane >> 4;

  const u32 asBase = (u32)(uintptr_t)&As[0][0][0];
  const u32 bsBase = (u32)(uintptr_t)&Bs[0][0][0];
  const int crow = tid >> 2;                   // 0..63 (+64 on second issue)
  const int ckc  = tid & 3;                    // 16B chunk within 64B k-row
  const u16* agp = A + (size_t)(m0 + crow) * lda + ckc * 8;
  const u16* bgp = B + (size_t)(n0 + crow) * ldb + ckc * 8;
  const size_t aStep = (size_t)64 * lda;
  const size_t bStep = (size_t)64 * ldb;
  const u32 aOff0 = asBase + crow * kRowBytes + ckc * 16;
  const u32 bOff0 = bsBase + crow * kRowBytes + ckc * 16;

#define STAGE_TILE(s, k0) do {                                   \
    async_cp16(aOff0 + (s) * kStageBytes, agp + (k0));           \
    async_cp16(aOff0 + 64 * kRowBytes + (s) * kStageBytes,       \
               agp + aStep + (k0));                              \
    async_cp16(bOff0 + (s) * kStageBytes, bgp + (k0));           \
    async_cp16(bOff0 + 64 * kRowBytes + (s) * kStageBytes,       \
               bgp + bStep + (k0));                              \
  } while (0)

  v8f acc[4][2] = {};
  const int nIter = Kd / BK;
  STAGE_TILE(0, 0);
  for (int t = 0; t < nIter; ++t) {
    const int s = t & 1;
    wait_async0();            // this wave's copies of tile t have landed
    __syncthreads();          // everyone's copies landed; prev reads of s^1 done
    if (t + 1 < nIter) STAGE_TILE(s ^ 1, (t + 1) * BK);

    FragU af[4], bfr[2];
#pragma unroll
    for (int i = 0; i < 4; ++i) {
      const uint4* p = (const uint4*)&As[s][wm * 64 + i * 16 + lm][lh * 8];
      af[i].q[0] = p[0];      // K 0..7   (this half)
      af[i].q[1] = p[2];      // K 16..23 (this half)
    }
#pragma unroll
    for (int j = 0; j < 2; ++j) {
      const uint4* p = (const uint4*)&Bs[s][wn * 32 + j * 16 + lm][lh * 8];
      bfr[j].q[0] = p[0];
      bfr[j].q[1] = p[2];
    }
#pragma unroll
    for (int i = 0; i < 4; ++i)
#pragma unroll
      for (int j = 0; j < 2; ++j)
        acc[i][j] = wmma_bf16(af[i].v, bfr[j].v, acc[i][j]);
  }
#undef STAGE_TILE

  float* C = outF32 ? (Cg + bb * sCb + hh * sCh) : nullptr;
  u16*  Cb = outBF ? (Cbg + bb * sCBb + hh * sCBh) : nullptr;
#pragma unroll
  for (int j = 0; j < 2; ++j) {
    const int nn = n0 + wn * 32 + j * 16 + lm;
    if (nn >= Ncols) continue;
    const float bval = bias ? bias[nn] : 0.f;
#pragma unroll
    for (int i = 0; i < 4; ++i) {
      const int mmb = m0 + wm * 64 + i * 16 + (lh << 3);
      float v[8];
#pragma unroll
      for (int g = 0; g < 8; ++g) {
        float x = acc[i][j][g] * scale + bval;
        if (doGelu) x = gelu_tanh(x);
        v[g] = x;
      }
      if (outF32) {
#pragma unroll
        for (int g = 0; g < 8; ++g) {
          const size_t idx = (size_t)(mmb + g) * ldc + nn;
          float x = v[g];
          if (doResid) x += C[idx];
          C[idx] = x;
        }
      }
      if (outBF) {
        if (storeT) {
          uint4 pk;
          pk.x = pack2(v[0], v[1]); pk.y = pack2(v[2], v[3]);
          pk.z = pack2(v[4], v[5]); pk.w = pack2(v[6], v[7]);
          *(uint4*)&Cb[(size_t)nn * ldcb + mmb] = pk;   // contiguous in m
        } else {
#pragma unroll
          for (int g = 0; g < 8; ++g)
            Cb[(size_t)(mmb + g) * ldcb + nn] = f2bf(v[g]);
        }
      }
    }
  }
}

// LayerNorm, one block per row (D=1024); bf16 output
__global__ __launch_bounds__(256) void ln_kernel(const float* __restrict__ x,
                                                 const float* __restrict__ w,
                                                 const float* __restrict__ b,
                                                 u16* __restrict__ out, int D)
{
  __shared__ float s1[256], s2[256];
  const size_t row = blockIdx.x;
  const float* xr = x + row * D;
  float sum = 0.f, sq = 0.f;
  for (int j = threadIdx.x; j < D; j += 256) { float v = xr[j]; sum += v; sq += v * v; }
  s1[threadIdx.x] = sum; s2[threadIdx.x] = sq;
  __syncthreads();
  for (int off = 128; off > 0; off >>= 1) {
    if ((int)threadIdx.x < off) { s1[threadIdx.x] += s1[threadIdx.x + off];
                                  s2[threadIdx.x] += s2[threadIdx.x + off]; }
    __syncthreads();
  }
  const float mean = s1[0] / (float)D;
  const float var  = s2[0] / (float)D - mean * mean;
  const float rstd = rsqrtf(var + 1e-5f);
  u16* orow = out + row * D;
  for (int j = threadIdx.x; j < D; j += 256)
    orow[j] = f2bf((xr[j] - mean) * rstd * w[j] + b[j]);
}

// softmax over rows of 256 f32 -> bf16 out
__global__ __launch_bounds__(256) void softmax256_kernel(const float* __restrict__ din,
                                                         u16* __restrict__ dout)
{
  __shared__ float s[256];
  const size_t row = blockIdx.x;
  const float v = din[row * 256 + threadIdx.x];
  s[threadIdx.x] = v;
  __syncthreads();
  for (int off = 128; off > 0; off >>= 1) {
    if ((int)threadIdx.x < off) s[threadIdx.x] = fmaxf(s[threadIdx.x], s[threadIdx.x + off]);
    __syncthreads();
  }
  const float mx = s[0];
  __syncthreads();
  const float e = __expf(v - mx);
  s[threadIdx.x] = e;
  __syncthreads();
  for (int off = 128; off > 0; off >>= 1) {
    if ((int)threadIdx.x < off) s[threadIdx.x] += s[threadIdx.x + off];
    __syncthreads();
  }
  dout[row * 256 + threadIdx.x] = f2bf(e / s[0]);
}

// flat f32 -> bf16
__global__ __launch_bounds__(256) void cvt_kernel(const float* __restrict__ s,
                                                  u16* __restrict__ d, ll n)
{
  ll i = ((ll)blockIdx.x * 256 + threadIdx.x) * 4;
  if (i + 4 <= n) {
    float4 f = *(const float4*)(s + i);
    uint2 o; o.x = pack2(f.x, f.y); o.y = pack2(f.z, f.w);
    *(uint2*)(d + i) = o;
  } else {
    for (; i < n; ++i) d[i] = f2bf(s[i]);
  }
}

// (R x C) f32 -> transposed (C x R) bf16
__global__ __launch_bounds__(256) void cvtT_kernel(const float* __restrict__ s,
                                                   u16* __restrict__ d, int R, int C)
{
  ll id = (ll)blockIdx.x * 256 + threadIdx.x;
  if (id < (ll)R * C) {
    int c = (int)(id / R), r = (int)(id - (ll)c * R);
    d[id] = f2bf(s[(ll)r * C + c]);
  }
}

// (R x C) f32 -> transposed+padded (PC x R) bf16, zero pad rows c >= C
__global__ __launch_bounds__(256) void cvtTpad_kernel(const float* __restrict__ s,
                                                      u16* __restrict__ d,
                                                      int R, int C, int PC)
{
  ll id = (ll)blockIdx.x * 256 + threadIdx.x;
  if (id < (ll)PC * R) {
    int c = (int)(id / R), r = (int)(id - (ll)c * R);
    d[id] = (c < C) ? f2bf(s[(ll)r * C + c]) : (u16)0;
  }
}

static void launch_gemm(hipStream_t st, const u16* A, const u16* B, const float* bias,
                        float* C, u16* Cb, int M, int Ncols, int Kd,
                        int lda, int ldb, int ldc, int ldcb,
                        float scale, int gelu, int resid, int outF32, int outBF, int storeT,
                        int batch, int zdiv,
                        ll sAb, ll sAh, ll sBb, ll sBh,
                        ll sCb, ll sCh, ll sCBb, ll sCBh)
{
  dim3 grid((Ncols + BN - 1) / BN, (M + BM - 1) / BM, batch);
  gemm_wmma_kernel<<<grid, 256, 0, st>>>(A, B, bias, C, Cb, Ncols, Kd, lda, ldb, ldc,
                                         ldcb, scale, gelu, resid, outF32, outBF, storeT,
                                         zdiv, sAb, sAh, sBb, sBh, sCb, sCh, sCBb, sCBh);
}
static void launch_cvtT(hipStream_t st, const float* s, u16* d, int R, int C) {
  ll n = (ll)R * C;
  cvtT_kernel<<<(unsigned)((n + 255) / 256), 256, 0, st>>>(s, d, R, C);
}

extern "C" void kernel_launch(void* const* d_in, const int* in_sizes, int n_in,
                              void* d_out, int out_size, void* d_ws, size_t ws_size,
                              hipStream_t stream) {
  (void)in_sizes; (void)n_in; (void)out_size; (void)ws_size;
  const int Bb = 8, N = 2048, D = 1024, KP = 256, DEPTH = 4, FF = 4096, OUT = 411;
  const int H = 8, dh = 128, OUTP = 512;
  const int M = Bb * N;                               // 16384
  const size_t MD = (size_t)M * D;
  const size_t BHNK = (size_t)Bb * H * N * KP;        // 33,554,432
  const float attn_scale = 0.08838834764831845f;      // dh^-0.5

  const float* in_x   = (const float*)d_in[0];
  const float* ln1_w  = (const float*)d_in[1];
  const float* ln1_b  = (const float*)d_in[2];
  const float* Wq_all = (const float*)d_in[3];
  const float* Wk_all = (const float*)d_in[4];
  const float* Wv_all = (const float*)d_in[5];
  const float* Pk_all = (const float*)d_in[6];
  const float* Pv_all = (const float*)d_in[7];
  const float* Wo_all = (const float*)d_in[8];
  const float* bo_all = (const float*)d_in[9];
  const float* ln2_w  = (const float*)d_in[10];
  const float* ln2_b  = (const float*)d_in[11];
  const float* W1_all = (const float*)d_in[12];
  const float* b1_all = (const float*)d_in[13];
  const float* W2_all = (const float*)d_in[14];
  const float* b2_all = (const float*)d_in[15];
  const float* fc_w   = (const float*)d_in[16];
  const float* fc_b   = (const float*)d_in[17];

  // ---- workspace carve-up ----
  float* xbuf = (float*)d_ws;                 // (B,N,D) f32 residual stream
  float* dots = xbuf + MD;                    // (B,H,N,KP) f32 attention logits
  u16* u = (u16*)(dots + BHNK);
  u16* hbf = u;  u += MD;                     // LN output (GEMM A)
  u16* qbf = u;  u += MD;                     // q  (aliased later by o)
  u16* tT  = u;  u += MD;                     // (h@Wk/Wv) transposed: (D x M)
  u16* kbf = u;  u += (size_t)Bb * KP * D;    // k (B,KP,D)
  u16* vT  = u;  u += (size_t)Bb * KP * D;    // v transposed (B,D,KP)
  u16* attn= u;  u += BHNK;                   // softmax out (B,H,N,KP)
  u16* ffh = u;  u += (size_t)M * FF;         // gelu(h@W1+b1) (M x FF)
  u16* xbf = u;  u += MD;                     // x in bf16 for final head
  u16* wq  = u;  u += (size_t)D * D;          // transposed weights (BT layout)
  u16* wk  = u;  u += (size_t)D * D;
  u16* wv  = u;  u += (size_t)D * D;
  u16* wo  = u;  u += (size_t)D * D;
  u16* w1  = u;  u += (size_t)D * FF;         // (FF x D)
  u16* w2  = u;  u += (size_t)D * FF;         // (D x FF)
  u16* pk  = u;  u += (size_t)KP * N;         // Pk^T (KP x N)
  u16* pv  = u;  u += (size_t)KP * N;
  u16* fcT = u;  u += (size_t)OUTP * D;       // fc_w^T padded (512 x D)
  u16* obf = qbf;                             // o aliases q

  hipMemcpyAsync(xbuf, in_x, MD * sizeof(float), hipMemcpyDeviceToDevice, stream);

  for (int i = 0; i < DEPTH; ++i) {
    // one-time f32 -> bf16 (transposed into BT layout) weight conversion
    launch_cvtT(stream, Wq_all + (size_t)i * D * D, wq, D, D);
    launch_cvtT(stream, Wk_all + (size_t)i * D * D, wk, D, D);
    launch_cvtT(stream, Wv_all + (size_t)i * D * D, wv, D, D);
    launch_cvtT(stream, Wo_all + (size_t)i * D * D, wo, D, D);
    launch_cvtT(stream, W1_all + (size_t)i * D * FF, w1, D, FF);
    launch_cvtT(stream, W2_all + (size_t)i * FF * D, w2, FF, D);
    launch_cvtT(stream, Pk_all + (size_t)i * N * KP, pk, N, KP);  // -> Pk^T (KP x N)
    launch_cvtT(stream, Pv_all + (size_t)i * N * KP, pv, N, KP);
    const float* bo = bo_all + (size_t)i * D;
    const float* b1 = b1_all + (size_t)i * FF;
    const float* b2 = b2_all + (size_t)i * D;

    // h = LN1(x)
    ln_kernel<<<M, 256, 0, stream>>>(xbuf, ln1_w + (size_t)i * D,
                                     ln1_b + (size_t)i * D, hbf, D);
    // q = h @ Wq                      (bf16 out)
    launch_gemm(stream, hbf, wq, nullptr, nullptr, qbf, M, D, D, D, D, 0, D,
                1.f, 0, 0, 0, 1, 0, 1, 1, 0,0, 0,0, 0,0, 0,0);
    // tT = (h @ Wk)^T  (D x M)
    launch_gemm(stream, hbf, wk, nullptr, nullptr, tT, M, D, D, D, D, 0, M,
                1.f, 0, 0, 0, 1, 1, 1, 1, 0,0, 0,0, 0,0, 0,0);
    // k[b] = Pk^T @ tmp[b]  -> (KP x D) bf16
    launch_gemm(stream, pk, tT, nullptr, nullptr, kbf, KP, D, N, N, M, 0, D,
                1.f, 0, 0, 0, 1, 0, Bb, 1,
                0,0, (ll)N,0, 0,0, (ll)KP * D, 0);
    // tT = (h @ Wv)^T ; v[b]^T = (Pv^T @ tmp[b])^T -> (D x KP) bf16
    launch_gemm(stream, hbf, wv, nullptr, nullptr, tT, M, D, D, D, D, 0, M,
                1.f, 0, 0, 0, 1, 1, 1, 1, 0,0, 0,0, 0,0, 0,0);
    launch_gemm(stream, pv, tT, nullptr, nullptr, vT, KP, D, N, N, M, 0, KP,
                1.f, 0, 0, 0, 1, 1, Bb, 1,
                0,0, (ll)N,0, 0,0, (ll)D * KP, 0);
    // dots[b,h] = scale * q_slice @ k_slice^T   (f32 out)
    launch_gemm(stream, qbf, kbf, nullptr, dots, nullptr, N, KP, dh, D, D, KP, 0,
                attn_scale, 0, 0, 1, 0, 0, Bb * H, H,
                (ll)N * D, (ll)dh, (ll)KP * D, (ll)dh,
                (ll)H * N * KP, (ll)N * KP, 0, 0);
    // softmax (f32 in, bf16 out)
    softmax256_kernel<<<(unsigned)(BHNK / KP), 256, 0, stream>>>(dots, attn);
    // o_slice = attn @ v_slice   (bf16 out into (B,N,D))
    launch_gemm(stream, attn, vT, nullptr, nullptr, obf, N, dh, KP, KP, KP, 0, D,
                1.f, 0, 0, 0, 1, 0, Bb * H, H,
                (ll)H * N * KP, (ll)N * KP, (ll)D * KP, (ll)dh * KP,
                0, 0, (ll)N * D, (ll)dh);
    // x = x + o @ Wo + bo  (f32, residual)
    launch_gemm(stream, obf, wo, bo, xbuf, nullptr, M, D, D, D, D, D, 0,
                1.f, 0, 1, 1, 0, 0, 1, 1, 0,0, 0,0, 0,0, 0,0);
    // h = LN2(x); ffh = gelu(h @ W1 + b1) (bf16); x = x + ffh @ W2 + b2
    ln_kernel<<<M, 256, 0, stream>>>(xbuf, ln2_w + (size_t)i * D,
                                     ln2_b + (size_t)i * D, hbf, D);
    launch_gemm(stream, hbf, w1, b1, nullptr, ffh, M, FF, D, D, D, 0, FF,
                1.f, 1, 0, 0, 1, 0, 1, 1, 0,0, 0,0, 0,0, 0,0);
    launch_gemm(stream, ffh, w2, b2, xbuf, nullptr, M, D, FF, FF, FF, D, 0,
                1.f, 0, 1, 1, 0, 0, 1, 1, 0,0, 0,0, 0,0, 0,0);
  }

  // final head: out = x @ fc_w + fc_b  (OUT=411, padded BT weight to 512 rows)
  cvt_kernel<<<(unsigned)((MD + 1023) / 1024), 256, 0, stream>>>(xbuf, xbf, (ll)MD);
  cvtTpad_kernel<<<(unsigned)(((ll)OUTP * D + 255) / 256), 256, 0, stream>>>(
      fc_w, fcT, D, OUT, OUTP);
  launch_gemm(stream, xbf, fcT, fc_b, (float*)d_out, nullptr, M, OUT, D, D, D, OUT, 0,
              1.f, 0, 0, 1, 0, 0, 1, 1, 0,0, 0,0, 0,0, 0,0);
}
